// StyleEncoder_70334384439492
// MI455X (gfx1250) — matrix operand order repaired and checked by reading, not compiled, not run on hardware
//
#include <hip/hip_runtime.h>
#include <hip/hip_bf16.h>
#include <math.h>

typedef __attribute__((ext_vector_type(16))) _Float16 v16h;
typedef __attribute__((ext_vector_type(8)))  _Float16 v8h;
typedef __attribute__((ext_vector_type(8)))  float    v8f;

// ---------------------------------------------------------------------------
// zero helper (accumulators re-zeroed every call: graph replay safe)
// ---------------------------------------------------------------------------
__global__ void zero_f32_kernel(float* __restrict__ p, int n) {
    int i = blockIdx.x * 256 + threadIdx.x;
    if (i < n) p[i] = 0.f;
}

// ---------------------------------------------------------------------------
// weight pack: OIHW f32  ->  [OC][k' = tap*IC + ic] f16  (tap-major K)
// ---------------------------------------------------------------------------
__global__ void pack_w_kernel(const float* __restrict__ src,
                              _Float16* __restrict__ dst,
                              int K, int IC, int icShift, int icMask, int n) {
    int idx = blockIdx.x * 256 + threadIdx.x;
    if (idx >= n) return;
    int oc = idx / K;
    int k  = idx - oc * K;
    int t  = k >> icShift;       // tap 0..8
    int ic = k & icMask;
    dst[idx] = (_Float16)src[((size_t)(oc * IC + ic)) * 9 + t];
}

// ---------------------------------------------------------------------------
// conv1: 3->32 reflect pad, f32 direct, NCHW input -> NHWC output
// ---------------------------------------------------------------------------
__global__ void conv1_kernel(const float* __restrict__ x,
                             const float* __restrict__ w,
                             const float* __restrict__ b,
                             float* __restrict__ y) {   // (8, 65536, 32) NHWC
    int idx = blockIdx.x * 256 + threadIdx.x;           // 8*65536*32
    int oc = idx & 31;
    int t = idx >> 5;
    int p = t & 65535;
    int n = t >> 16;
    int oy = p >> 8, ox = p & 255;
    float acc = b[oc];
    #pragma unroll
    for (int ic = 0; ic < 3; ++ic)
        #pragma unroll
        for (int r = 0; r < 3; ++r) {
            int iy = oy + r - 1;
            iy = iy < 0 ? -iy : (iy > 255 ? 510 - iy : iy);
            #pragma unroll
            for (int s = 0; s < 3; ++s) {
                int ix = ox + s - 1;
                ix = ix < 0 ? -ix : (ix > 255 ? 510 - ix : ix);
                acc += x[((n * 3 + ic) * 256 + iy) * 256 + ix] *
                       w[((oc * 3 + ic) * 3 + r) * 3 + s];
            }
        }
    y[idx] = acc;
}

// ---------------------------------------------------------------------------
// IN stats stage A: coalesced sweep, LDS ds_add_f32 partials per channel,
// then global_atomic_add_f32 into per-(n,c) accumulators.
// ---------------------------------------------------------------------------
__global__ void in_stats_accum_kernel(const float* __restrict__ x,
                                      float* __restrict__ gsum,
                                      float* __restrict__ gsum2,
                                      int C, int cMask, int pcShift) {
    extern __shared__ float lds[];      // 2*C floats
    float* ls  = lds;
    float* ls2 = lds + C;
    for (int i = threadIdx.x; i < 2 * C; i += 256) lds[i] = 0.f;
    __syncthreads();
    size_t base = (size_t)blockIdx.x * 16384;
    int n = (int)(base >> pcShift);
    for (int i = 0; i < 64; ++i) {
        size_t idx = base + i * 256 + threadIdx.x;
        float v = x[idx];
        int c = (int)idx & cMask;
        atomicAdd(&ls[c], v);
        atomicAdd(&ls2[c], v * v);
    }
    __syncthreads();
    for (int c = threadIdx.x; c < C; c += 256) {
        atomicAdd(&gsum[n * C + c], ls[c]);
        atomicAdd(&gsum2[n * C + c], ls2[c]);
    }
}

__global__ void in_stats_final_kernel(const float* __restrict__ gsum,
                                      const float* __restrict__ gsum2,
                                      float* __restrict__ mean,
                                      float* __restrict__ rstd,
                                      int NC, float invHW) {
    int nc = blockIdx.x * 256 + threadIdx.x;
    if (nc >= NC) return;
    float m = gsum[nc] * invHW;
    float v = gsum2[nc] * invHW - m * m;
    mean[nc] = m;
    rstd[nc] = rsqrtf(v + 1e-5f);
}

// ---------------------------------------------------------------------------
// normalize (+optional lrelu), NHWC f32 -> NHWC f16
// ---------------------------------------------------------------------------
__global__ void norm_act_kernel(const float* __restrict__ x,
                                const float* __restrict__ mean,
                                const float* __restrict__ rstd,
                                _Float16* __restrict__ y,
                                int C, int cMask, int pcShift,
                                int leaky, int total) {
    int idx = blockIdx.x * 256 + threadIdx.x;
    if (idx >= total) return;
    int c = idx & cMask;
    int n = idx >> pcShift;
    int nc = n * C + c;
    float v = (x[idx] - mean[nc]) * rstd[nc];
    if (leaky && v < 0.f) v *= 0.2f;
    y[idx] = (_Float16)v;
}

// ---------------------------------------------------------------------------
// Implicit-GEMM conv via V_WMMA_F32_16X16X32_F16, NHWC activations,
// tap-major packed weights [OC][tap*IC+ic]; all geometry compile-time.
//   MODE 0: zero pad 1, stride STRIDE       (conv2, conv3)
//   MODE 1: reflect pad 1, stride 1         (conv5)
//   MODE 2: lhs-dilation 2, pad low 1       (conv_transpose)
// 2x2 register blocking per wave: 2 oc-tiles x 2 pixel-tiles, 4 accumulators;
// per K-step: 2 A-frags + 2 B-frags -> 4 WMMAs (2.0 b128 loads per wmma).
// ---------------------------------------------------------------------------
template <int MODE, int IC, int IH, int IW, int OC, int OWSHIFT, int STRIDE,
          int ACT, int P>
__global__ void __launch_bounds__(128)
conv_wmma_kernel(const _Float16* __restrict__ in,   // (N, IH, IW, IC) f16
                 const _Float16* __restrict__ wp,   // (OC, 9*IC) f16 tap-major
                 const float* __restrict__ bias,
                 float* __restrict__ out) {         // (N, OH, OW, OC) f32
    constexpr int OW = 1 << OWSHIFT;
    constexpr int K  = IC * 9;

    const int lane = threadIdx.x & 31;
    const int wave = threadIdx.x >> 5;
    const int n    = blockIdx.z;
    const int oc0  = blockIdx.y << 5;                 // pair of oc tiles
    const int p0   = ((blockIdx.x << 2) + wave) << 5; // pair of pixel tiles

    const int g  = lane >> 4;
    const int ln = lane & 15;

    const _Float16* arow0 = wp + (size_t)(oc0 + ln) * K + (g << 4);
    const _Float16* arow1 = arow0 + (size_t)16 * K;

    const int pA = p0 + ln;
    const int pB = pA + 16;
    const int oyA = pA >> OWSHIFT, oxA = pA & (OW - 1);
    const int oyB = pB >> OWSHIFT, oxB = pB & (OW - 1);
    const _Float16* inN = in + (size_t)n * IH * IW * IC;

    v8f acc00 = {}, acc01 = {}, acc10 = {}, acc11 = {};

    #pragma unroll
    for (int t = 0; t < 9; ++t) {
        const int r = t / 3, s = t % 3;       // compile-time after unroll

        auto tap = [&](int oy, int ox, bool& valid) -> const _Float16* {
            if (MODE == 0) {
                int iy = oy * STRIDE - 1 + r;
                int ix = ox * STRIDE - 1 + s;
                valid = (iy >= 0) & (iy < IH) & (ix >= 0) & (ix < IW);
                return inN + ((size_t)iy * IW + ix) * IC;
            } else if (MODE == 1) {
                int iy = oy + r - 1;
                iy = iy < 0 ? -iy : (iy >= IH ? 2 * IH - 2 - iy : iy);
                int ix = ox + s - 1;
                ix = ix < 0 ? -ix : (ix >= IW ? 2 * IW - 2 - ix : ix);
                valid = true;
                return inN + ((size_t)iy * IW + ix) * IC;
            } else {          // lhs_dilation=2, low pad 1 (both dims)
                int iyd = oy + r - 1;
                int ixd = ox + s - 1;
                valid = (iyd >= 0) & (ixd >= 0) & !(iyd & 1) & !(ixd & 1) &&
                        ((iyd >> 1) < IH) && ((ixd >> 1) < IW);
                return inN + ((size_t)(iyd >> 1) * IW + (ixd >> 1)) * IC;
            }
        };
        bool vA, vB;
        const _Float16* bpA = tap(oyA, oxA, vA);
        const _Float16* bpB = tap(oyB, oxB, vB);

        #pragma unroll
        for (int c0 = 0; c0 < IC; c0 += 32) {
            const int k0 = t * IC + c0;
            // ---- A fragments: constant-offset b128 loads
            v16h a0, a1;
            {
                const v8h* ap0 = (const v8h*)(arow0 + k0);
                const v8h* ap1 = (const v8h*)(arow1 + k0);
                v8h a0l = ap0[0], a0h = ap0[1];
                v8h a1l = ap1[0], a1h = ap1[1];
                #pragma unroll
                for (int i = 0; i < 8; ++i) {
                    a0[i] = a0l[i]; a0[8 + i] = a0h[i];
                    a1[i] = a1l[i]; a1[8 + i] = a1h[i];
                }
            }
            // ---- B fragments: 16 contiguous channels each
            v16h b0, b1;
            if (vA) {
                const v8h* bp = (const v8h*)(bpA + c0 + (g << 4));
                v8h x0 = bp[0], x1 = bp[1];
                #pragma unroll
                for (int i = 0; i < 8; ++i) { b0[i] = x0[i]; b0[8 + i] = x1[i]; }
            } else {
                #pragma unroll
                for (int i = 0; i < 16; ++i) b0[i] = (_Float16)0.f;
            }
            if (vB) {
                const v8h* bp = (const v8h*)(bpB + c0 + (g << 4));
                v8h x0 = bp[0], x1 = bp[1];
                #pragma unroll
                for (int i = 0; i < 8; ++i) { b1[i] = x0[i]; b1[8 + i] = x1[i]; }
            } else {
                #pragma unroll
                for (int i = 0; i < 16; ++i) b1[i] = (_Float16)0.f;
            }

            acc00 = __builtin_amdgcn_wmma_f32_16x16x32_f16(
                false, a0, false, b0, (short)0, acc00, false, false);
            acc01 = __builtin_amdgcn_wmma_f32_16x16x32_f16(
                false, a0, false, b1, (short)0, acc01, false, false);
            acc10 = __builtin_amdgcn_wmma_f32_16x16x32_f16(
                false, a1, false, b0, (short)0, acc10, false, false);
            acc11 = __builtin_amdgcn_wmma_f32_16x16x32_f16(
                false, a1, false, b1, (short)0, acc11, false, false);
        }
    }

    // ---- epilogue: bias (+tanh); lane holds 8 consecutive oc per pixel
    auto store_tile = [&](v8f& acc, int ocb, int p) {
        float* obase = out + ((size_t)n * P + p) * OC + ocb + (g << 3);
        float v[8];
        #pragma unroll
        for (int r2 = 0; r2 < 8; ++r2) {
            float t2 = acc[r2] + bias[ocb + (g << 3) + r2];
            v[r2] = ACT ? tanhf(t2) : t2;
        }
        float4* o4 = (float4*)obase;
        o4[0] = make_float4(v[0], v[1], v[2], v[3]);
        o4[1] = make_float4(v[4], v[5], v[6], v[7]);
    };
    store_tile(acc00, oc0,      pA);
    store_tile(acc01, oc0,      pB);
    store_tile(acc10, oc0 + 16, pA);
    store_tile(acc11, oc0 + 16, pB);
}

// ---------------------------------------------------------------------------
// per-image class pixel counts on downsampled labels (::2, ::2)
// ---------------------------------------------------------------------------
__global__ void seg_counts_kernel(const int* __restrict__ seg,
                                  float* __restrict__ counts) {
    int b = blockIdx.x;
    __shared__ int c[19];
    if (threadIdx.x < 19) c[threadIdx.x] = 0;
    __syncthreads();
    for (int p = threadIdx.x; p < 16384; p += 256) {
        int y = p >> 7, x = p & 127;
        int l = seg[(b * 256 + 2 * y) * 256 + 2 * x];
        atomicAdd(&c[l], 1);
    }
    __syncthreads();
    if (threadIdx.x < 19) counts[b * 19 + threadIdx.x] = (float)c[threadIdx.x];
}

// ---------------------------------------------------------------------------
// masked pooling over NHWC `up`: 512 channel-threads, label broadcast,
// conflict-free LDS [19][512] partials, global f32 atomics into sums
// ---------------------------------------------------------------------------
__global__ void seg_accum_kernel(const float* __restrict__ up,   // (8,16384,512)
                                 const int* __restrict__ seg,
                                 float* __restrict__ sums) {     // (8,19,512)
    int b = blockIdx.y;
    int tid = threadIdx.x;                 // channel
    __shared__ float part[19][512];
    #pragma unroll
    for (int s = 0; s < 19; ++s) part[s][tid] = 0.f;
    __syncthreads();
    int p0 = blockIdx.x * 1024;
    for (int i = 0; i < 1024; ++i) {
        int p = p0 + i;
        int y = p >> 7, x = p & 127;
        int l = seg[(b * 256 + 2 * y) * 256 + 2 * x];   // broadcast
        float v = up[((size_t)b * 16384 + p) * 512 + tid];
        part[l][tid] += v;
    }
    __syncthreads();
    #pragma unroll
    for (int s = 0; s < 19; ++s)
        atomicAdd(&sums[((size_t)b * 19 + s) * 512 + tid], part[s][tid]);
}

__global__ void style_final_kernel(const float* __restrict__ sums,
                                   const float* __restrict__ counts,
                                   float* __restrict__ out) {
    int idx = blockIdx.x * 256 + threadIdx.x;      // 8*19*512
    if (idx >= 8 * 19 * 512) return;
    int bs = idx >> 9;
    float cnt = counts[bs];
    out[idx] = cnt > 0.f ? sums[idx] / fmaxf(cnt, 1.f) : 0.f;
}

// ---------------------------------------------------------------------------
extern "C" void kernel_launch(void* const* d_in, const int* in_sizes, int n_in,
                              void* d_out, int out_size, void* d_ws,
                              size_t ws_size, hipStream_t stream) {
    (void)in_sizes; (void)n_in; (void)out_size; (void)ws_size;

    const float* x   = (const float*)d_in[0];
    const int*   seg = (const int*)d_in[1];
    const float* w1  = (const float*)d_in[2];
    const float* b1  = (const float*)d_in[3];
    const float* w2  = (const float*)d_in[4];
    const float* b2  = (const float*)d_in[5];
    const float* w3  = (const float*)d_in[6];
    const float* b3  = (const float*)d_in[7];
    const float* wt  = (const float*)d_in[8];
    const float* bt  = (const float*)d_in[9];
    const float* w5  = (const float*)d_in[10];
    const float* b5  = (const float*)d_in[11];
    float* out = (float*)d_out;

    char* ws = (char*)d_ws;
    size_t off = 0;
    auto carve = [&](size_t bytes) -> char* {
        char* p = ws + off;
        off = (off + bytes + 255) & ~(size_t)255;
        return p;
    };
    float*    F32   = (float*)carve((size_t)8 * 512 * 128 * 128 * 4);  // NHWC conv out (reused)
    _Float16* H16a  = (_Float16*)carve((size_t)8 * 256 * 128 * 128 * 2);
    _Float16* H16b  = (_Float16*)carve((size_t)8 * 256 * 128 * 128 * 2);
    float*    gsum  = (float*)carve(4096 * 4);
    float*    gsum2 = (float*)carve(4096 * 4);      // contiguous after gsum
    float*    meanB = (float*)carve(4096 * 4);
    float*    rstdB = (float*)carve(4096 * 4);
    _Float16* w2h   = (_Float16*)carve((size_t)64  * 288  * 2);
    _Float16* w3h   = (_Float16*)carve((size_t)128 * 576  * 2);
    _Float16* wth   = (_Float16*)carve((size_t)256 * 1152 * 2);
    _Float16* w5h   = (_Float16*)carve((size_t)512 * 2304 * 2);
    float*    sums  = (float*)carve((size_t)8 * 19 * 512 * 4);
    float*    counts= (float*)carve(8 * 19 * 4);

    // pack weights to tap-major f16
    auto pack = [&](const float* src, _Float16* dst, int IC, int OC, int icShift) {
        int K = IC * 9, n = OC * K;
        pack_w_kernel<<<(n + 255) / 256, 256, 0, stream>>>(
            src, dst, K, IC, icShift, IC - 1, n);
    };
    pack(w2, w2h, 32, 64, 5);
    pack(w3, w3h, 64, 128, 6);
    pack(wt, wth, 128, 256, 7);
    pack(w5, w5h, 256, 512, 8);

    auto stats = [&](const float* src, int C, int cMask, int pcShift,
                     int total, int NC, float invHW) {
        zero_f32_kernel<<<(2 * 4096 + 255) / 256, 256, 0, stream>>>(gsum, 2 * 4096);
        in_stats_accum_kernel<<<total / 16384, 256, 2 * C * 4, stream>>>(
            src, gsum, gsum2, C, cMask, pcShift);
        in_stats_final_kernel<<<(NC + 255) / 256, 256, 0, stream>>>(
            gsum, gsum2, meanB, rstdB, NC, invHW);
    };

    // ---- layer 1: conv 3->32 reflect + IN + lrelu -> H16a (8,256,256,32)
    conv1_kernel<<<(8 * 65536 * 32) / 256, 256, 0, stream>>>(x, w1, b1, F32);
    stats(F32, 32, 31, 21, 8 * 65536 * 32, 8 * 32, 1.f / 65536.f);
    norm_act_kernel<<<(8 * 65536 * 32) / 256, 256, 0, stream>>>(
        F32, meanB, rstdB, H16a, 32, 31, 21, 1, 8 * 65536 * 32);

    // ---- layer 2: conv 32->64 s2 (WMMA) + IN + lrelu -> H16b (8,128,128,64)
    conv_wmma_kernel<0, 32, 256, 256, 64, 7, 2, 0, 16384>
        <<<dim3(128, 2, 8), 128, 0, stream>>>(H16a, w2h, b2, F32);
    stats(F32, 64, 63, 20, 8 * 16384 * 64, 8 * 64, 1.f / 16384.f);
    norm_act_kernel<<<(8 * 16384 * 64) / 256, 256, 0, stream>>>(
        F32, meanB, rstdB, H16b, 64, 63, 20, 1, 8 * 16384 * 64);

    // ---- layer 3: conv 64->128 s2 (WMMA) + IN + lrelu -> H16a (8,64,64,128)
    conv_wmma_kernel<0, 64, 128, 128, 128, 6, 2, 0, 4096>
        <<<dim3(32, 4, 8), 128, 0, stream>>>(H16b, w3h, b3, F32);
    stats(F32, 128, 127, 19, 8 * 4096 * 128, 8 * 128, 1.f / 4096.f);
    norm_act_kernel<<<(8 * 4096 * 128) / 256, 256, 0, stream>>>(
        F32, meanB, rstdB, H16a, 128, 127, 19, 1, 8 * 4096 * 128);

    // ---- layer 4: conv_transpose 128->256 s2 (WMMA) + IN -> H16b (8,128,128,256)
    conv_wmma_kernel<2, 128, 64, 64, 256, 7, 1, 0, 16384>
        <<<dim3(128, 8, 8), 128, 0, stream>>>(H16a, wth, bt, F32);
    stats(F32, 256, 255, 22, 8 * 16384 * 256, 8 * 256, 1.f / 16384.f);
    norm_act_kernel<<<(8 * 16384 * 256) / 256, 256, 0, stream>>>(
        F32, meanB, rstdB, H16b, 256, 255, 22, 0, 8 * 16384 * 256);

    // ---- layer 5: conv 256->512 reflect (WMMA) + tanh fused -> F32 (8,128,128,512)
    conv_wmma_kernel<1, 256, 128, 128, 512, 7, 1, 1, 16384>
        <<<dim3(128, 16, 8), 128, 0, stream>>>(H16b, w5h, b5, F32);

    // ---- masked per-class pooling -> (8,19,512)
    seg_counts_kernel<<<8, 256, 0, stream>>>(seg, counts);
    zero_f32_kernel<<<(8 * 19 * 512 + 255) / 256, 256, 0, stream>>>(sums, 8 * 19 * 512);
    seg_accum_kernel<<<dim3(16, 8), 512, 0, stream>>>(F32, seg, sums);
    style_final_kernel<<<(8 * 19 * 512 + 255) / 256, 256, 0, stream>>>(
        sums, counts, out);
}
